// PAM_29042568856107
// MI455X (gfx1250) — compile-verified, matrix operand order checked by reading, not run on hardware
//
#include <hip/hip_runtime.h>
#include <hip/hip_bf16.h>

// ---------------------------------------------------------------------------
// PAM (position attention) fused kernels for MI455X / gfx1250.
//   x: (4, 256, 64, 64) f32 ; Wb,Wc: (32,256); Wd: (256,256); biases; gamma.
//   out = gamma * (V @ softmax(QK^T)^T) + x   (flash-attention fused)
// Precision: bf16 WMMA 16x16x32 with f32 accumulation.
// K/V tiles streamed into LDS by the Tensor Data Mover (TENSORcnt-tracked,
// double buffered, DMA overlapped with WMMA compute).
// ---------------------------------------------------------------------------

#define BATCH 4
#define CCH   256
#define CKD   32
#define NPIX  4096

typedef __bf16 bf16_t;
typedef __attribute__((ext_vector_type(16))) __bf16 v16bf;
typedef __attribute__((ext_vector_type(8)))  float  v8f;
typedef __attribute__((ext_vector_type(4)))  unsigned int u32x4;
typedef __attribute__((ext_vector_type(4)))  int i32x4;
typedef __attribute__((ext_vector_type(8)))  int i32x8;

// Load one WMMA 16-bit operand half-row pair from a contiguous 32-element
// (bf16) row. Lane-group g=0 takes elements [0..7] and [16..23];
// g=1 takes [8..15] and [24..31] (matches A and B operand layouts).
__device__ __forceinline__ v16bf load_op32(const bf16_t* rowp, int g) {
    v16bf r;
    const int o = g ? 8 : 0;
    ((uint4*)&r)[0] = *(const uint4*)(rowp + o);
    ((uint4*)&r)[1] = *(const uint4*)(rowp + 16 + o);
    return r;
}

__device__ __forceinline__ v8f wmma_bf16(v16bf a, v16bf b, v8f c) {
    return __builtin_amdgcn_wmma_f32_16x16x32_bf16(
        /*neg_a=*/false, a, /*neg_b=*/false, b,
        /*c_mod=*/(short)0, c, /*reuse_a=*/false, /*reuse_b=*/false);
}

// ---------------------------------------------------------------------------
// Tensor Data Mover descriptor helpers (CDNA5 ISA ch.8).
// 2D tile of 2-byte elements; LDS padding of 4 DWORDs (8 bf16) inserted every
// 16 DWORDs (one 32-element row) -> 40-element LDS row stride, bank-conflict
// free b128 reads.
// ---------------------------------------------------------------------------
__device__ __forceinline__ i32x8 tdm_group1_2d(unsigned tensor_d0, unsigned tensor_d1,
                                               unsigned tile_d0, unsigned tile_d1,
                                               unsigned long long stride0) {
    i32x8 g1;
    // workgroup_mask=0, data_size=1 (2B), pad_enable=1,
    // pad_interval=3 (16 DWORDs), pad_amount=3 (4 DWORDs)
    g1[0] = (int)((1u << 16) | (1u << 20) | (3u << 22) | (3u << 25));
    g1[1] = (int)((tensor_d0 & 0xFFFFu) << 16);                       // dim0[15:0]
    g1[2] = (int)((tensor_d0 >> 16) | ((tensor_d1 & 0xFFFFu) << 16)); // dim0[31:16], dim1[15:0]
    g1[3] = (int)((tensor_d1 >> 16) | (tile_d0 << 16));               // dim1[31:16], tile_d0
    g1[4] = (int)tile_d1;                                             // tile_d1, tile_d2=0
    g1[5] = (int)(unsigned)(stride0 & 0xFFFFFFFFu);                   // stride0[31:0]
    g1[6] = (int)(unsigned)(stride0 >> 32);                           // stride0[47:32]
    g1[7] = 0;
    return g1;
}

__device__ __forceinline__ void tdm_load2d(unsigned lds_addr, const void* gptr, i32x8 g1) {
    const unsigned long long ga = (unsigned long long)(uintptr_t)gptr;
    u32x4 g0;
    g0[0] = 1u;                                    // count=1, user descriptor
    g0[1] = lds_addr;                              // LDS byte address
    g0[2] = (unsigned)ga;                          // global_addr[31:0]
    g0[3] = (unsigned)(ga >> 32) | (2u << 30);     // global_addr[56:32], type=2
    const i32x4 z4 = {0, 0, 0, 0};
#if defined(__clang_major__) && (__clang_major__ >= 23)
    const i32x8 z8 = {0, 0, 0, 0, 0, 0, 0, 0};
    __builtin_amdgcn_tensor_load_to_lds(g0, g1, z4, z4, z8, 0);
#else
    __builtin_amdgcn_tensor_load_to_lds(g0, g1, z4, z4, 0);
#endif
}

// ---------------------------------------------------------------------------
// Kernel 0: convert Wd (256x256 f32) to bf16.
// ---------------------------------------------------------------------------
__global__ __launch_bounds__(256) void cvt_wd_kernel(const float* __restrict__ Wd,
                                                     bf16_t* __restrict__ Wdbf) {
    int i = blockIdx.x * blockDim.x + threadIdx.x;
    if (i < CCH * CCH) Wdbf[i] = (bf16_t)Wd[i];
}

// ---------------------------------------------------------------------------
// Kernel 1: Q/K projections. One thread per (batch, pixel).
// ---------------------------------------------------------------------------
__global__ __launch_bounds__(256) void qkproj_kernel(
    const float* __restrict__ x,
    const float* __restrict__ Wb, const float* __restrict__ bb,
    const float* __restrict__ Wc, const float* __restrict__ bc,
    bf16_t* __restrict__ Q, bf16_t* __restrict__ K) {
    const int gid = blockIdx.x * blockDim.x + threadIdx.x;   // 0 .. B*N-1
    const int b   = gid >> 12;
    const int n   = gid & (NPIX - 1);

    float qa[CKD], ka[CKD];
#pragma unroll
    for (int k = 0; k < CKD; ++k) { qa[k] = bb[k]; ka[k] = bc[k]; }

    const float* xb = x + (size_t)b * CCH * NPIX + n;
    for (int c = 0; c < CCH; ++c) {
        const float xv = xb[(size_t)c * NPIX];
#pragma unroll
        for (int k = 0; k < CKD; ++k) {
            qa[k] = fmaf(Wb[k * CCH + c], xv, qa[k]);
            ka[k] = fmaf(Wc[k * CCH + c], xv, ka[k]);
        }
    }
    bf16_t* qout = Q + (size_t)gid * CKD;
    bf16_t* kout = K + (size_t)gid * CKD;
#pragma unroll
    for (int k = 0; k < CKD; ++k) { qout[k] = (bf16_t)qa[k]; kout[k] = (bf16_t)ka[k]; }
}

// ---------------------------------------------------------------------------
// Kernel 2: V projection via WMMA.  V^T[b][c][n] = sum_cc Wd[c][cc]*x[b][cc][n]+bd[c]
// ---------------------------------------------------------------------------
__global__ __launch_bounds__(256) void vproj_kernel(
    const float* __restrict__ x, const bf16_t* __restrict__ Wdbf,
    const float* __restrict__ bd, bf16_t* __restrict__ V) {
    const int wave = threadIdx.x >> 5;
    const int lane = threadIdx.x & 31;
    const int lrow = lane & 15;
    const int g    = lane >> 4;
    const int wid  = blockIdx.x * (blockDim.x >> 5) + wave;  // 0 .. B*(N/16)-1
    const int b    = wid >> 8;                               // 256 tiles per batch
    const int n0   = (wid & 255) * 16;

    v8f acc[16];
#pragma unroll
    for (int t = 0; t < 16; ++t) acc[t] = (v8f){0.f,0.f,0.f,0.f,0.f,0.f,0.f,0.f};

    const float* xb = x + (size_t)b * CCH * NPIX;
    for (int cc = 0; cc < CCH; cc += 32) {
        v16bf bop;
#pragma unroll
        for (int i = 0; i < 8; ++i)
            bop[i] = (bf16_t)xb[(size_t)(cc + g * 8 + i) * NPIX + n0 + lrow];
#pragma unroll
        for (int i = 0; i < 8; ++i)
            bop[8 + i] = (bf16_t)xb[(size_t)(cc + 16 + g * 8 + i) * NPIX + n0 + lrow];
#pragma unroll
        for (int t = 0; t < 16; ++t) {
            const bf16_t* arow = Wdbf + (size_t)(t * 16 + lrow) * CCH + cc;
            v16bf aop = load_op32(arow, g);
            acc[t] = wmma_bf16(aop, bop, acc[t]);
        }
    }
#pragma unroll
    for (int t = 0; t < 16; ++t) {
#pragma unroll
        for (int r = 0; r < 8; ++r) {
            const int ch = t * 16 + g * 8 + r;
            V[((size_t)b * CCH + ch) * NPIX + n0 + lrow] = (bf16_t)(acc[t][r] + bd[ch]);
        }
    }
}

// ---------------------------------------------------------------------------
// Kernel 3: fused flash attention + residual, TDM-fed.
// Block = 8 waves = 128 queries of one batch. Per 32-key block (double-buffered
// in LDS, filled by tensor_load_to_lds issued from wave 0):
//   S^T = [K_blk0; K_blk1] x Q^T       (2x WMMA, D layout == keys x queries)
//   online softmax over keys (per-lane VGPR reduce + shfl_xor(16))
//   P^T packs straight into the B-operand layout (no cross-lane shuffles)
//   O^T[t] += V_tile(t) x P^T          (16x WMMA)
// Epilogue: out = gamma/l * O^T + x.
// ---------------------------------------------------------------------------
__global__ __launch_bounds__(256) void attn_kernel(
    const bf16_t* __restrict__ Q, const bf16_t* __restrict__ K,
    const bf16_t* __restrict__ V, const float* __restrict__ x,
    const float* __restrict__ gamma, float* __restrict__ out) {
    // Double-buffered, TDM-padded LDS tiles (40 bf16 = 80B row stride).
    __shared__ bf16_t Klds[2][32][40];
    __shared__ bf16_t Vlds[2][256][40];

    const int tid  = threadIdx.x;
    const int wave = tid >> 5;
    const int lane = tid & 31;
    const int lrow = lane & 15;
    const int g    = lane >> 4;
    const int b    = blockIdx.x >> 5;                 // 32 blocks per batch
    const int qbase = (blockIdx.x & 31) * 128 + wave * 16;

    const bf16_t* Kb = K + (size_t)b * NPIX * CKD;
    const bf16_t* Vb = V + (size_t)b * CCH * NPIX;

    // Q tile as B operand (held for the whole kernel).
    const v16bf qop = load_op32(Q + ((size_t)b * NPIX + qbase + lrow) * CKD, g);

    v8f acc[16];
#pragma unroll
    for (int t = 0; t < 16; ++t) acc[t] = (v8f){0.f,0.f,0.f,0.f,0.f,0.f,0.f,0.f};
    float m = -INFINITY, l = 0.f;

    // TDM descriptors: K tensor is (N x 32) row-major (tile = 32 rows x 32),
    // V tensor is (C x N) row-major (tile = 256 rows x 32 cols at column j).
    const i32x8 g1K = tdm_group1_2d(/*d0=*/CKD,  /*d1=*/NPIX, /*t0=*/32, /*t1=*/32,  /*stride0=*/CKD);
    const i32x8 g1V = tdm_group1_2d(/*d0=*/NPIX, /*d1=*/CCH,  /*t0=*/32, /*t1=*/CCH, /*stride0=*/NPIX);
    const unsigned kbase0 = (unsigned)(uintptr_t)&Klds[0][0][0];
    const unsigned kbase1 = (unsigned)(uintptr_t)&Klds[1][0][0];
    const unsigned vbase0 = (unsigned)(uintptr_t)&Vlds[0][0][0];
    const unsigned vbase1 = (unsigned)(uintptr_t)&Vlds[1][0][0];

    if (wave == 0) {                           // prologue: fill buffer 0
        tdm_load2d(kbase0, Kb, g1K);
        tdm_load2d(vbase0, Vb, g1V);
    }

    const int nblk = NPIX / 32;
    for (int jb = 0; jb < nblk; ++jb) {
        const int cur = jb & 1;
        if (wave == 0) __builtin_amdgcn_s_wait_tensorcnt(0);
        __syncthreads();                       // buf[cur] visible to all waves

        if (jb + 1 < nblk && wave == 0) {      // DMA next block into buf[cur^1]
            tdm_load2d(cur ? kbase0 : kbase1, Kb + (size_t)(jb + 1) * 32 * CKD, g1K);
            tdm_load2d(cur ? vbase0 : vbase1, Vb + (size_t)(jb + 1) * 32,       g1V);
        }

        // S^T = K_blk x Q^T  (two 16x16 tiles: keys j..j+15 and j+16..j+31)
        const v16bf ka0 = load_op32(&Klds[cur][lrow][0], g);
        const v16bf ka1 = load_op32(&Klds[cur][16 + lrow][0], g);
        const v8f zero = (v8f){0.f,0.f,0.f,0.f,0.f,0.f,0.f,0.f};
        v8f st0 = wmma_bf16(ka0, qop, zero);
        v8f st1 = wmma_bf16(ka1, qop, zero);

        // Online softmax over the 32 keys of this block (per query column).
        float tm = st0[0];
#pragma unroll
        for (int r = 1; r < 8; ++r) tm = fmaxf(tm, st0[r]);
#pragma unroll
        for (int r = 0; r < 8; ++r) tm = fmaxf(tm, st1[r]);
        tm = fmaxf(tm, __shfl_xor(tm, 16, 32));
        const float mnew = fmaxf(m, tm);
        const float corr = __expf(m - mnew);

        v16bf pt;
        float s = 0.f;
#pragma unroll
        for (int r = 0; r < 8; ++r) {
            const float p = __expf(st0[r] - mnew);
            s += p; pt[r] = (bf16_t)p;
        }
#pragma unroll
        for (int r = 0; r < 8; ++r) {
            const float p = __expf(st1[r] - mnew);
            s += p; pt[8 + r] = (bf16_t)p;
        }
        s += __shfl_xor(s, 16, 32);
        l = l * corr + s;
        m = mnew;

        if (__any(corr != 1.0f)) {             // uniform branch per wave
#pragma unroll
            for (int t = 0; t < 16; ++t)
#pragma unroll
                for (int r = 0; r < 8; ++r) acc[t][r] *= corr;
        }

        // O^T[t] += V_tile(t) x P^T
#pragma unroll
        for (int t = 0; t < 16; ++t) {
            const v16bf va = load_op32(&Vlds[cur][t * 16 + lrow][0], g);
            acc[t] = wmma_bf16(va, pt, acc[t]);
        }
        __syncthreads();                       // reads done before buf reuse
    }

    const float scale = gamma[0] / l;
#pragma unroll
    for (int t = 0; t < 16; ++t) {
#pragma unroll
        for (int r = 0; r < 8; ++r) {
            const int ch = t * 16 + g * 8 + r;
            const size_t idx = ((size_t)b * CCH + ch) * NPIX + qbase + lrow;
            out[idx] = fmaf(acc[t][r], scale, x[idx]);
        }
    }
}

// ---------------------------------------------------------------------------
// Launch
// ---------------------------------------------------------------------------
extern "C" void kernel_launch(void* const* d_in, const int* in_sizes, int n_in,
                              void* d_out, int out_size, void* d_ws, size_t ws_size,
                              hipStream_t stream) {
    const float* x     = (const float*)d_in[0];
    const float* Wb    = (const float*)d_in[1];
    const float* bb    = (const float*)d_in[2];
    const float* Wc    = (const float*)d_in[3];
    const float* bc    = (const float*)d_in[4];
    const float* Wd    = (const float*)d_in[5];
    const float* bd    = (const float*)d_in[6];
    const float* gamma = (const float*)d_in[7];
    float* out = (float*)d_out;

    char* ws = (char*)d_ws;
    bf16_t* Wdbf = (bf16_t*)(ws);                                  // 128 KB
    bf16_t* Qb   = (bf16_t*)(ws + 131072);                         // 1 MB
    bf16_t* Kb   = (bf16_t*)(ws + 131072 + 1048576);               // 1 MB
    bf16_t* Vb   = (bf16_t*)(ws + 131072 + 2 * 1048576);           // 8 MB

    cvt_wd_kernel<<<(CCH * CCH) / 256, 256, 0, stream>>>(Wd, Wdbf);
    qkproj_kernel<<<(BATCH * NPIX) / 256, 256, 0, stream>>>(x, Wb, bb, Wc, bc, Qb, Kb);
    vproj_kernel<<<(BATCH * (NPIX / 16)) / 8, 256, 0, stream>>>(x, Wdbf, bd, Vb);
    attn_kernel<<<BATCH * (NPIX / 128), 256, 0, stream>>>(Qb, Kb, Vb, x, gamma, out);
}
